// ScannedRNN_9491877724481
// MI455X (gfx1250) — matrix operand order repaired and checked by reading, not compiled, not run on hardware
//
#include <hip/hip_runtime.h>
#include <hip/hip_bf16.h>

// ---------------------------------------------------------------------------
// Scanned GRU (flax GRUCell with reset-to-zero carries) for MI455X / gfx1250.
//
// Phase 1 (parallel): gi = x @ Wi + bi for ALL timesteps at once.
//   One big bf16 WMMA GEMM: (T*B, D) @ (D, 3H) -> fp32 gi in workspace.
// Phase 2 (sequential): persistent kernel, 1024 steps, one device-scope grid
//   barrier per step, double-buffered carry. gh = h @ Wh via v_wmma bf16,
//   fp32 accumulate; Wh held entirely in VGPRs (loop-invariant); gi stream
//   prefetched across the barrier. Each workgroup owns a 32x32 tile of the
//   new carry and computes all 3 gates for its columns (no cross-WG gate mix).
// ---------------------------------------------------------------------------

#define TT  1024
#define BB  256
#define DD  256
#define HH  256
#define G3H 768
#define NWG 64          // persistent workgroups for the scan (8x8 tiles of 32x32)

typedef __bf16 bf16;
typedef __attribute__((ext_vector_type(16))) __bf16 bf16x16;
typedef __attribute__((ext_vector_type(8)))  __bf16 bf16x8;
typedef __attribute__((ext_vector_type(8)))  float  f32x8;
typedef __attribute__((ext_vector_type(4)))  float  f32x4;

union ABf { bf16x16 v; bf16x8 h[2]; };

// ---- WMMA fragment helpers (wave32 layouts per CDNA5 ISA 7.12.2) ----------

// A matrix 16x32 bf16, row-major source (ld = row stride in elements).
// lane 0-15: row m=lane, K chunks {0..7, 16..23}; lane 16-31: {8..15, 24..31}.
__device__ __forceinline__ bf16x16 load_a_frag(const bf16* base, int ld, int lane) {
    int m    = lane & 15;
    int koff = (lane >> 4) << 3;          // 0 or 8
    const bf16* p = base + m * ld + koff;
    ABf u;
    u.h[0] = *(const bf16x8*)(p);
    u.h[1] = *(const bf16x8*)(p + 16);
    return u.v;
}

// Same A fragment but converting fp32 -> bf16 on the fly (phase-1 only).
__device__ __forceinline__ bf16x16 load_a_frag_f32(const float* base, int ld, int lane) {
    int m    = lane & 15;
    int koff = (lane >> 4) << 3;
    const float* p = base + (long)m * ld + koff;
    f32x4 a0 = *(const f32x4*)(p);
    f32x4 a1 = *(const f32x4*)(p + 4);
    f32x4 a2 = *(const f32x4*)(p + 16);
    f32x4 a3 = *(const f32x4*)(p + 20);
    bf16x8 lo, hi;
    #pragma unroll
    for (int i = 0; i < 4; i++) {
        lo[i]     = (bf16)a0[i];  lo[i + 4] = (bf16)a1[i];
        hi[i]     = (bf16)a2[i];  hi[i + 4] = (bf16)a3[i];
    }
    ABf u; u.h[0] = lo; u.h[1] = hi;
    return u.v;
}

// B matrix 32x16 bf16 from an N-major (pre-transposed) weight: row n holds the
// K-contiguous column n of the original weight. lane 0-15: n=lane, K {0..7, 16..23};
// lane 16-31: n=lane-16, K {8..15, 24..31}.
__device__ __forceinline__ bf16x16 load_b_frag(const bf16* baseN, int ld, int lane) {
    int n    = lane & 15;
    int koff = (lane >> 4) << 3;          // 0 or 8
    const bf16* p = baseN + n * ld + koff;
    ABf u;
    u.h[0] = *(const bf16x8*)(p);
    u.h[1] = *(const bf16x8*)(p + 16);
    return u.v;
}

__device__ __forceinline__ f32x8 wmma_bf16(bf16x16 a, bf16x16 b, f32x8 c) {
    return __builtin_amdgcn_wmma_f32_16x16x32_bf16(
        /*neg_a=*/false, a, /*neg_b=*/false, b,
        /*c_mod=*/(short)0, c, /*reuse_a=*/false, /*reuse_b=*/false);
}

__device__ __forceinline__ float sigmoidf(float x) {
    return 1.0f / (1.0f + __expf(-x));
}

// ---- setup kernels --------------------------------------------------------

// Transpose Wi, Wh (D x 3H row-major) into N-major bf16 (3H x D), K-contiguous.
__global__ __launch_bounds__(256) void pack_weights(
    const float* __restrict__ Wi, const float* __restrict__ Wh,
    bf16* __restrict__ WiT, bf16* __restrict__ WhT) {
    int i = blockIdx.x * 256 + threadIdx.x;
    if (i < G3H * DD) {
        int n = i / DD;            // 0..767 : output column of original weight
        int k = i % DD;            // 0..255 : K index, contiguous in output
        WiT[i] = (bf16)Wi[(long)k * G3H + n];
        WhT[i] = (bf16)Wh[(long)k * G3H + n];
    }
}

// Initialize carry buffer 0 with reset-masked init_carry; zero barrier counter.
// (resets is a raw jax bool array: 1 byte per element)
__global__ __launch_bounds__(256) void init_state(
    const float* __restrict__ h0, const unsigned char* __restrict__ resets,
    float* __restrict__ hfp, bf16* __restrict__ hbf,
    unsigned int* __restrict__ counter) {
    int i = blockIdx.x * 256 + threadIdx.x;
    if (i == 0) *counter = 0u;
    if (i < BB * HH) {
        int b = i / HH;
        float h = resets[b] ? 0.0f : h0[i];     // reset at t=0 applies to init carry
        hfp[i] = h;
        hbf[i] = (bf16)h;
    }
}

// ---- Phase 1: gi = x @ Wi + bi for all T*B rows ---------------------------
// Block = 256 threads (8 waves). Block tile 128 rows x 64 cols; each wave
// computes a 16x64 strip (4 accumulators), K = 256 in steps of 32.
__global__ __launch_bounds__(256) void gi_gemm(
    const float* __restrict__ x,      // (T*B, D) fp32
    const bf16*  __restrict__ WiT,    // (3H, D)  bf16, N-major
    const float* __restrict__ bi,     // (3H,)
    float* __restrict__ gi) {         // (T*B, 3H) fp32
    const int lane = threadIdx.x & 31;
    const int wave = threadIdx.x >> 5;                 // 0..7
    const long m0  = (long)blockIdx.x * 128 + wave * 16;
    const int  n0  = blockIdx.y * 64;

    f32x8 acc[4] = {};
    const float* arow = x + m0 * DD;
    for (int k = 0; k < DD; k += 32) {
        bf16x16 a = load_a_frag_f32(arow + k, DD, lane);
        #pragma unroll
        for (int j = 0; j < 4; j++) {
            bf16x16 b = load_b_frag(WiT + (n0 + j * 16) * DD + k, DD, lane);
            acc[j] = wmma_bf16(a, b, acc[j]);
        }
    }
    // C/D layout: n = lane&15 ; row = (lane>=16 ? 8 : 0) + i
    const int n  = lane & 15;
    const int mb = (lane >> 4) << 3;
    #pragma unroll
    for (int j = 0; j < 4; j++) {
        int col  = n0 + j * 16 + n;
        float bv = bi[col];
        float* o = gi + (m0 + mb) * G3H + col;
        #pragma unroll
        for (int i = 0; i < 8; i++)
            __builtin_nontemporal_store(acc[j][i] + bv, o + (long)i * G3H);
    }
}

// ---- Phase 2: persistent sequential scan ----------------------------------
// 64 workgroups x 128 threads (4 waves). WG owns a 32x32 tile of new_h and
// computes r/z/n gate GEMMs for its columns. One grid barrier per timestep;
// carry is double-buffered so a single barrier suffices. Wh fragments are
// preloaded into VGPRs (loop-invariant); gi loads are pipelined one step
// ahead so their HBM latency hides behind the barrier spin.
__global__ __launch_bounds__(128) void gru_scan(
    const float* __restrict__ gi,            // (T*B, 3H) fp32
    const unsigned char* __restrict__ resets,// (T, B) bool (1 byte)
    const bf16*  __restrict__ WhT,           // (3H, H) bf16 N-major
    const float* __restrict__ bhn,           // (H,)
    float* __restrict__ hfp,                 // [2][B*H] fp32 carry (post-reset)
    bf16*  __restrict__ hbf,                 // [2][B*H] bf16 carry (post-reset)
    unsigned int* __restrict__ counter,      // grid barrier
    float* __restrict__ out) {               // (T, B, H) fp32
    const int lane = threadIdx.x & 31;
    const int wave = threadIdx.x >> 5;                // 0..3
    const int bid  = blockIdx.x;                      // 0..63
    const int m0   = ((bid >> 3) << 5) + ((wave >> 1) << 4);  // row of 16x16 tile
    const int n0   = ((bid & 7) << 5) + ((wave & 1) << 4);    // col of 16x16 tile
    const int nn   = lane & 15;
    const int mb   = (lane >> 4) << 3;
    const float bhn_j = bhn[n0 + nn];
    const int j = n0 + nn;                            // output column of this lane

    // ---- preload all Wh fragments: 3 gates x 8 k-steps = 192 VGPRs --------
    bf16x16 bw[3][8];
    #pragma unroll
    for (int kk = 0; kk < 8; kk++) {
        #pragma unroll
        for (int g = 0; g < 3; g++)
            bw[g][kk] = load_b_frag(WhT + (g * HH + n0) * HH + kk * 32, HH, lane);
    }

    // ---- prefetch gi for t = 0 --------------------------------------------
    float gr[8], gz[8], gn[8];
    #pragma unroll
    for (int i = 0; i < 8; i++) {
        const float* gip = gi + (long)(m0 + mb + i) * G3H;
        gr[i] = __builtin_nontemporal_load(gip + j);
        gz[i] = __builtin_nontemporal_load(gip + HH + j);
        gn[i] = __builtin_nontemporal_load(gip + 2 * HH + j);
    }

    for (int t = 0; t < TT; t++) {
        const int buf = t & 1;
        const bf16*  hb = hbf + buf * (BB * HH);
        const float* hf = hfp + buf * (BB * HH);

        // ---- gh = h @ Wh for the 3 gate column-blocks ----
        f32x8 ar = {}, az = {}, an = {};
        const bf16* hrow = hb + m0 * HH;
        #pragma unroll
        for (int kk = 0; kk < 8; kk++) {
            bf16x16 a = load_a_frag(hrow + kk * 32, HH, lane);
            ar = wmma_bf16(a, bw[0][kk], ar);
            az = wmma_bf16(a, bw[1][kk], az);
            an = wmma_bf16(a, bw[2][kk], an);
        }

        // ---- prefetch gi for t+1 (independent of recurrence; hides HBM
        //      latency behind the elementwise math + barrier spin) ----
        float pr[8], pz[8], pn[8];
        if (t + 1 < TT) {
            const long tbn = (long)(t + 1) * BB;
            #pragma unroll
            for (int i = 0; i < 8; i++) {
                const float* gip = gi + (tbn + m0 + mb + i) * G3H;
                pr[i] = __builtin_nontemporal_load(gip + j);
                pz[i] = __builtin_nontemporal_load(gip + HH + j);
                pn[i] = __builtin_nontemporal_load(gip + 2 * HH + j);
            }
        }

        // ---- elementwise GRU gating ----
        const long tb = (long)t * BB;
        #pragma unroll
        for (int i = 0; i < 8; i++) {
            const int b = m0 + mb + i;
            float r  = sigmoidf(gr[i] + ar[i]);
            float z  = sigmoidf(gz[i] + az[i]);
            float nv = tanhf(gn[i] + r * (an[i] + bhn_j));
            float hold = hf[b * HH + j];
            float hnew = (1.0f - z) * nv + z * hold;
            __builtin_nontemporal_store(hnew, out + (tb + b) * HH + j);
            if (t + 1 < TT) {
                // Apply NEXT step's reset at write time: carry buffers hold
                // the post-reset h that step t+1 consumes (GEMM and z*h term).
                float hw = resets[(t + 1) * BB + b] ? 0.0f : hnew;
                const int nb = (t + 1) & 1;
                hfp[nb * (BB * HH) + b * HH + j] = hw;
                hbf[nb * (BB * HH) + b * HH + j] = (bf16)hw;
            }
        }

        if (t + 1 < TT) {
            // ---- device-scope grid barrier (one per timestep) ----
            __threadfence();
            __syncthreads();
            if (threadIdx.x == 0) {
                __hip_atomic_fetch_add(counter, 1u, __ATOMIC_ACQ_REL,
                                       __HIP_MEMORY_SCOPE_AGENT);
                const unsigned target = (unsigned)(t + 1) * NWG;
                while (__hip_atomic_load(counter, __ATOMIC_ACQUIRE,
                                         __HIP_MEMORY_SCOPE_AGENT) < target) {
                    __builtin_amdgcn_s_sleep(2);
                }
            }
            __syncthreads();
            __threadfence();

            // rotate prefetched gi registers
            #pragma unroll
            for (int i = 0; i < 8; i++) { gr[i] = pr[i]; gz[i] = pz[i]; gn[i] = pn[i]; }
        }
    }
}

// ---------------------------------------------------------------------------

extern "C" void kernel_launch(void* const* d_in, const int* in_sizes, int n_in,
                              void* d_out, int out_size, void* d_ws, size_t ws_size,
                              hipStream_t stream) {
    const float*         init_carry = (const float*)d_in[0];         // (B, H)
    const float*         ins        = (const float*)d_in[1];         // (T, B, D)
    const unsigned char* resets     = (const unsigned char*)d_in[2]; // (T, B) bool
    const float*         Wi         = (const float*)d_in[3];         // (D, 3H)
    const float*         bi         = (const float*)d_in[4];         // (3H,)
    const float*         Wh         = (const float*)d_in[5];         // (H, 3H)
    const float*         bhn        = (const float*)d_in[6];         // (H,)
    float* out = (float*)d_out;                                      // (T, B, H)

    char* ws = (char*)d_ws;
    size_t off = 0;
    float* gi  = (float*)(ws + off); off += (size_t)TT * BB * G3H * sizeof(float);
    bf16*  WiT = (bf16*) (ws + off); off += (size_t)G3H * DD * sizeof(bf16);
    bf16*  WhT = (bf16*) (ws + off); off += (size_t)G3H * HH * sizeof(bf16);
    float* hfp = (float*)(ws + off); off += 2ull * BB * HH * sizeof(float);
    bf16*  hbf = (bf16*) (ws + off); off += 2ull * BB * HH * sizeof(bf16);
    unsigned int* counter = (unsigned int*)(ws + off); off += 256;
    (void)ws_size; (void)in_sizes; (void)n_in; (void)out_size;

    pack_weights<<<(G3H * DD + 255) / 256, 256, 0, stream>>>(Wi, Wh, WiT, WhT);
    init_state  <<<(BB * HH + 255) / 256, 256, 0, stream>>>(init_carry, resets,
                                                            hfp, hbf, counter);
    dim3 g1(TT * BB / 128, G3H / 64);
    gi_gemm<<<g1, 256, 0, stream>>>(ins, WiT, bi, gi);
    gru_scan<<<NWG, 128, 0, stream>>>(gi, resets, WhT, bhn, hfp, hbf, counter, out);
}
